// AdditiveAttention_81784767250745
// MI455X (gfx1250) — matrix operand order repaired and checked by reading, not compiled, not run on hardware
//
#include <hip/hip_runtime.h>

#define HH 1024
#define SS 2048
#define BB 32
#define MT 128         // keys rows per workgroup in score kernel (8 waves x 16)
#define NCH 128        // N (output-o) chunk per accumulator set (8 wmma tiles)
#define LDA 1032       // padded LDS row stride for A (halfs): 2064B -> 4-bank shift/row
#define LDB 40         // padded LDS row stride for B (halfs): 80B  -> conflict-free

typedef __attribute__((ext_vector_type(16))) __bf16 v16bf;
typedef __attribute__((ext_vector_type(8)))  float  v8f;

__device__ __forceinline__ unsigned short f2bf(float f) {
  unsigned int u = __float_as_uint(f);
  u += 0x7FFFu + ((u >> 16) & 1u);      // round-to-nearest-even
  return (unsigned short)(u >> 16);
}

// branch-free tanh: exact saturation for |x|>9.5 in fp32; no EXEC divergence
__device__ __forceinline__ float fast_tanh(float x) {
  float cx = fminf(fmaxf(x, -9.5f), 9.5f);
  float t  = __expf(2.0f * cx);
  return __fdividef(t - 1.0f, t + 1.0f);
}

// ---- 1) convert U_w fp32 -> bf16 workspace ----
__global__ void cvt_uw_kernel(const float* __restrict__ Uw, unsigned short* __restrict__ out) {
  int i = (blockIdx.x * blockDim.x + threadIdx.x) * 4;
  float4 f = *(const float4*)(Uw + i);
  uint2 p;
  p.x = (unsigned int)f2bf(f.x) | ((unsigned int)f2bf(f.y) << 16);
  p.y = (unsigned int)f2bf(f.z) | ((unsigned int)f2bf(f.w) << 16);
  *(uint2*)(out + i) = p;
}

// ---- 2) WqTot[b][o] = query[b]·W_w[o] + W_b[o] + U_b[o] ----
__global__ void wq_kernel(const float* __restrict__ query, const float* __restrict__ Ww,
                          const float* __restrict__ Wb, const float* __restrict__ Ub,
                          float* __restrict__ WqTot) {
  int idx = blockIdx.x * blockDim.x + threadIdx.x;   // 0..32767
  int b = idx >> 10, o = idx & (HH - 1);
  const float4* q = (const float4*)(query + (size_t)b * HH);
  const float4* w = (const float4*)(Ww + (size_t)o * HH);
  float acc = Wb[o] + Ub[o];
#pragma unroll 4
  for (int i = 0; i < HH / 4; ++i) {
    float4 qa = q[i], wa = w[i];
    acc += qa.x * wa.x + qa.y * wa.y + qa.z * wa.z + qa.w * wa.w;
  }
  WqTot[idx] = acc;
}

// ---- 3) fused: scores[m] = sum_o v[o] * tanh(WqTot[b][o] + keys[m]·Uw[o]) ----
__global__ void __launch_bounds__(256)
score_kernel(const float* __restrict__ keys, const unsigned short* __restrict__ Uwb,
             const float* __restrict__ WqTot, const float* __restrict__ vw,
             float* __restrict__ scores) {
  extern __shared__ unsigned short sm[];
  unsigned short* lA  = sm;                        // [MT][LDA] bf16 keys tile
  unsigned short* lB0 = sm + MT * LDA;             // [NCH][LDB] B tile buffer 0
  unsigned short* lB1 = lB0 + NCH * LDB;           // [NCH][LDB] B tile buffer 1

  const int tid  = threadIdx.x;          // 0..255 (8 waves)
  const int wave = tid >> 5;
  const int lane = tid & 31;
  const int m0   = blockIdx.x * MT;
  const int b    = m0 >> 11;             // all MT rows share one batch (128 | 2048)

  // Phase 1: stage keys[m0:m0+MT][0:1024] -> bf16 LDS (converted once)
  const float* kbase = keys + (size_t)m0 * HH;
  for (int i = tid; i < MT * HH / 4; i += 256) {
    float4 f = ((const float4*)kbase)[i];
    int e = i * 4;
    int r = e >> 10, c = e & (HH - 1);
    uint2 p;
    p.x = (unsigned int)f2bf(f.x) | ((unsigned int)f2bf(f.y) << 16);
    p.y = (unsigned int)f2bf(f.z) | ((unsigned int)f2bf(f.w) << 16);
    *(uint2*)&lA[r * LDA + c] = p;
  }

  const int lane15 = lane & 31 & 15;
  const int lhA = (lane >> 4) << 3;      // A: K offset 0 / 8 per ISA 16-bit A layout
  const int lhB = (lane >> 4) << 3;      // B: K offset 0 / 8 (same layout as A)
  const int row = wave * 16 + lane15;    // this lane's A row within tile

  // B staging assignment: each thread moves 16 halfs of the 128x32 tile
  const int ol = tid >> 1;               // o-row within tile
  const int kh = (tid & 1) << 4;         // k-offset 0/16

  float s[8];
#pragma unroll
  for (int v = 0; v < 8; ++v) s[v] = 0.f;

  for (int o0 = 0; o0 < HH; o0 += NCH) {
    v8f acc[NCH / 16];
#pragma unroll
    for (int t = 0; t < NCH / 16; ++t)
#pragma unroll
      for (int v = 0; v < 8; ++v) acc[t][v] = 0.f;

    const unsigned short* srcB = Uwb + (size_t)(o0 + ol) * HH + kh;

    // prologue: stage k-tile 0 into lB0 (last reader of lB0 was >=1 barrier ago)
    {
      uint4 p0 = *(const uint4*)srcB;
      uint4 p1 = *(const uint4*)(srcB + 8);
      *(uint4*)&lB0[ol * LDB + kh]     = p0;
      *(uint4*)&lB0[ol * LDB + kh + 8] = p1;
    }
    __syncthreads();   // also covers phase-1 A staging on first o0

    for (int ki = 0; ki < HH / 32; ++ki) {
      const int k0 = ki * 32;
      unsigned short* cur = (ki & 1) ? lB1 : lB0;
      unsigned short* nxt = (ki & 1) ? lB0 : lB1;

      // issue next tile's global loads early (latency overlaps compute below)
      uint4 n0, n1;
      const bool more = (ki + 1) < (HH / 32);
      if (more) {
        const unsigned short* src = srcB + k0 + 32;
        n0 = *(const uint4*)src;
        n1 = *(const uint4*)(src + 8);
      }

      v16bf a;
      { // A operand: elements e<8 -> K=k0+lhA+e ; e>=8 -> K=k0+16+lhA+(e-8)
        const unsigned short* ap = &lA[row * LDA + k0 + lhA];
        union { uint4 q[2]; v16bf v; } u;
        u.q[0] = *(const uint4*)ap;
        u.q[1] = *(const uint4*)(ap + 16);
        a = u.v;
      }

      // load all 8 B operands into distinct registers, then issue 8 WMMAs
      v16bf bb[NCH / 16];
#pragma unroll
      for (int t = 0; t < NCH / 16; ++t) {
        const unsigned short* bp = &cur[(t * 16 + lane15) * LDB + lhB];
        union { uint4 q[2]; v16bf v; } ub;
        ub.q[0] = *(const uint4*)bp;
        ub.q[1] = *(const uint4*)(bp + 16);
        bb[t] = ub.v;
      }
#pragma unroll
      for (int t = 0; t < NCH / 16; ++t)
        acc[t] = __builtin_amdgcn_wmma_f32_16x16x32_bf16(
            false, a, false, bb[t], (short)0, acc[t], false, false);

      if (more) {
        *(uint4*)&nxt[ol * LDB + kh]     = n0;
        *(uint4*)&nxt[ol * LDB + kh + 8] = n1;
      }
      __syncthreads();   // nxt staged for ki+1; cur free for rewrite at ki+2
    }

    // epilogue: tanh + dot with v, accumulate per-lane partial score
#pragma unroll
    for (int t = 0; t < NCH / 16; ++t) {
      int o = o0 + t * 16 + lane15;
      float wq = WqTot[b * HH + o];
      float vv = vw[o];
#pragma unroll
      for (int v = 0; v < 8; ++v)
        s[v] += fast_tanh(acc[t][v] + wq) * vv;
    }
  }

  // reduce over the 16 lanes sharing each C/D row (C/D: lanes 0-15 rows 0-7, 16-31 rows 8-15)
#pragma unroll
  for (int v = 0; v < 8; ++v) {
    float t = s[v];
    t += __shfl_xor(t, 1, 16);
    t += __shfl_xor(t, 2, 16);
    t += __shfl_xor(t, 4, 16);
    t += __shfl_xor(t, 8, 16);
    s[v] = t;
  }
  int gm = m0 + wave * 16;
#pragma unroll
  for (int v = 0; v < 8; ++v) {
    if (lane == v)      scores[gm + v]     = s[v];
    if (lane == 16 + v) scores[gm + 8 + v] = s[v];
  }
}

// ---- 4) softmax over S per batch row (v_b cancels exactly) ----
__global__ void softmax_kernel(const float* __restrict__ scores, float* __restrict__ attn) {
  __shared__ float sh[SS];
  __shared__ float red[8];
  int b = blockIdx.x, tid = threadIdx.x;
  int wave = tid >> 5, lane = tid & 31;
  float m = -3.4e38f;
  for (int i = tid; i < SS; i += 256) {
    float v = scores[b * SS + i];
    sh[i] = v;
    m = fmaxf(m, v);
  }
  for (int k = 1; k < 32; k <<= 1) m = fmaxf(m, __shfl_xor(m, k, 32));
  if (lane == 0) red[wave] = m;
  __syncthreads();
  m = red[0];
#pragma unroll
  for (int w = 1; w < 8; ++w) m = fmaxf(m, red[w]);
  __syncthreads();
  float sum = 0.f;
  for (int i = tid; i < SS; i += 256) {
    float e = __expf(sh[i] - m);
    sh[i] = e;
    sum += e;
  }
  for (int k = 1; k < 32; k <<= 1) sum += __shfl_xor(sum, k, 32);
  if (lane == 0) red[wave] = sum;
  __syncthreads();
  sum = 0.f;
#pragma unroll
  for (int w = 0; w < 8; ++w) sum += red[w];
  float inv = 1.0f / sum;
  for (int i = tid; i < SS; i += 256)
    attn[b * SS + i] = sh[i] * inv;
}

// ---- 5) context = attn · keys ----
__global__ void zero_kernel(float* __restrict__ p) {
  p[blockIdx.x * blockDim.x + threadIdx.x] = 0.f;
}

__global__ void ctx_kernel(const float* __restrict__ keys, const float* __restrict__ attn,
                           float* __restrict__ ctx) {
  int b  = blockIdx.x >> 4;
  int sc = blockIdx.x & 15;
  int h0 = threadIdx.x * 4;
  float4 acc = {0.f, 0.f, 0.f, 0.f};
  int sbase = sc * (SS / 16);
  for (int s = 0; s < SS / 16; ++s) {
    float a = attn[b * SS + sbase + s];
    float4 k = *(const float4*)(keys + ((size_t)(b * SS + sbase + s)) * HH + h0);
    acc.x += a * k.x; acc.y += a * k.y; acc.z += a * k.z; acc.w += a * k.w;
  }
  float* d = ctx + b * HH + h0;
  atomicAdd(d + 0, acc.x);
  atomicAdd(d + 1, acc.y);
  atomicAdd(d + 2, acc.z);
  atomicAdd(d + 3, acc.w);
}

extern "C" void kernel_launch(void* const* d_in, const int* in_sizes, int n_in,
                              void* d_out, int out_size, void* d_ws, size_t ws_size,
                              hipStream_t stream) {
  const float* query = (const float*)d_in[0];
  const float* keys  = (const float*)d_in[1];
  const float* Ww    = (const float*)d_in[2];
  const float* Wb    = (const float*)d_in[3];
  const float* Uw    = (const float*)d_in[4];
  const float* Ub    = (const float*)d_in[5];
  const float* vw    = (const float*)d_in[6];
  // d_in[7] = v_b : shifts all scores equally -> cancels in softmax

  unsigned short* Uwb = (unsigned short*)d_ws;                         // 2 MB
  float* WqTot  = (float*)((char*)d_ws + (size_t)HH * HH * 2);         // 128 KB
  float* scores = WqTot + BB * HH;                                     // 256 KB

  float* ctx  = (float*)d_out;           // [B,H]
  float* attn = ctx + BB * HH;           // [B,S]

  cvt_uw_kernel<<<HH * HH / (256 * 4), 256, 0, stream>>>(Uw, Uwb);
  wq_kernel<<<BB * HH / 256, 256, 0, stream>>>(query, Ww, Wb, Ub, WqTot);

  size_t smem = (size_t)(MT * LDA + 2 * NCH * LDB) * sizeof(unsigned short); // ~284 KB
  (void)hipFuncSetAttribute((const void*)score_kernel,
                            hipFuncAttributeMaxDynamicSharedMemorySize, (int)smem);
  score_kernel<<<BB * SS / MT, 256, smem, stream>>>(keys, Uwb, WqTot, vw, scores);

  softmax_kernel<<<BB, 256, 0, stream>>>(scores, attn);
  zero_kernel<<<BB * HH / 256, 256, 0, stream>>>(ctx);
  ctx_kernel<<<BB * 16, 256, 0, stream>>>(keys, attn, ctx);
}
